// FracAttention_53480932770123
// MI455X (gfx1250) — matrix operand order repaired and checked by reading, not compiled, not run on hardware
//
#include <hip/hip_runtime.h>
#include <stdint.h>

// ---------------- types ----------------
typedef __attribute__((ext_vector_type(16))) __bf16        v16bf;
typedef __attribute__((ext_vector_type(8)))  float         v8f;
typedef __attribute__((ext_vector_type(4)))  unsigned int  v4u;
typedef __attribute__((ext_vector_type(4)))  float         fv4;
typedef int gv4i __attribute__((vector_size(16)));   // matches builtin param type

#define DM    2048
#define NH    16
#define DKH   128
#define BATCH 2
#define SEQ   2048
#define MTOK  (BATCH * SEQ)   // 4096 tokens

// ---------------- CDNA5 async global->LDS path (probe via __has_builtin) ----
#if defined(__has_builtin)
#  if __has_builtin(__builtin_amdgcn_global_load_async_to_lds_b128)
#    define HAVE_ASYNC 1
#  endif
#endif
#ifndef HAVE_ASYNC
#  define HAVE_ASYNC 0
#endif

#if HAVE_ASYNC
#  define ASYNC_CP16(g, l)                                                     \
      __builtin_amdgcn_global_load_async_to_lds_b128(                          \
          (__attribute__((address_space(1))) gv4i*)(g),                        \
          (__attribute__((address_space(3))) gv4i*)(l), 0, 0)
#  if __has_builtin(__builtin_amdgcn_s_wait_asynccnt)
#    define ASYNC_WAIT() __builtin_amdgcn_s_wait_asynccnt(0)
#  else
#    define ASYNC_WAIT() asm volatile("s_wait_asynccnt 0x0" ::: "memory")
#  endif
#else
#  define ASYNC_CP16(g, l) (*(v4u*)(l) = *(const v4u*)(g))
#  define ASYNC_WAIT()
#endif

union Frag16 { v4u u[2]; v16bf v; };

// load 16 bf16 as two 16B chunks (A-fragment style: split K ranges)
__device__ __forceinline__ v16bf ld_frag2(const __bf16* p0, const __bf16* p1) {
    Frag16 f;
    f.u[0] = *reinterpret_cast<const v4u*>(p0);
    f.u[1] = *reinterpret_cast<const v4u*>(p1);
    return f.v;
}
// load 16 contiguous bf16 (B-fragment style: 32B strip along K)
__device__ __forceinline__ v16bf ld_frag32(const __bf16* p) {
    return ld_frag2(p, p + 8);
}

__device__ __forceinline__ v8f wmma_bf16(v16bf a, v16bf b, v8f c) {
    return __builtin_amdgcn_wmma_f32_16x16x32_bf16(
        /*neg_a=*/false, a, /*neg_b=*/false, b,
        /*c_mod=*/(short)0, c, /*reuse_a=*/false, /*reuse_b=*/false);
}

// convert 16 fp32 (4x fv4 in regs) -> 16 bf16 in LDS
__device__ __forceinline__ void cvt_store16(__bf16* d, const fv4* r) {
#pragma unroll
    for (int i = 0; i < 4; ++i) {
        d[4 * i + 0] = (__bf16)r[i].x;
        d[4 * i + 1] = (__bf16)r[i].y;
        d[4 * i + 2] = (__bf16)r[i].z;
        d[4 * i + 3] = (__bf16)r[i].w;
    }
}

// ============================================================================
// GEMM: C[M,N] = A[M,K] (fp32) @ W[N,K]^T (fp32, torch Linear layout) + bias[N]
// OUT_MODE 0: bf16 row-major [M,N]
// OUT_MODE 1: bf16 per-head transposed Vt[((b*NH+h)*DKH + d)*SEQ + s]
// OUT_MODE 2: fp32 row-major [M,N]
// Block tile 128x128, K-step 32, 8 waves (2m x 4n, each 64x32). Software
// pipeline: next k-tile rides in VGPRs while WMMAs consume the LDS tile.
// ============================================================================
template <int OUT_MODE>
__global__ __launch_bounds__(256) void gemm_xwT_kernel(
    const float* __restrict__ A, const float* __restrict__ W,
    const float* __restrict__ bias, void* __restrict__ out,
    int M, int N, int K)
{
    __shared__ __bf16 As[128 * 40];   // padded stride 40 (80B) -> conflict-free b128
    __shared__ __bf16 Ws[128 * 40];

    const int tid  = threadIdx.x;
    const int lane = tid & 31;
    const int wave = tid >> 5;
    const int ln   = lane & 15;
    const int half = lane >> 4;
    const int wm   = wave & 1;        // 0..1  -> 64-row slice
    const int wn   = wave >> 1;       // 0..3  -> 32-col slice
    const int bm   = blockIdx.y * 128;
    const int bn   = blockIdx.x * 128;

    const int srow = tid >> 1;          // 0..127
    const int scol = (tid & 1) * 16;    // 0 or 16

    const float* gA = A + (size_t)(bm + srow) * K + scol;
    const float* gW = W + (size_t)(bn + srow) * K + scol;

    v8f acc[4][2];
#pragma unroll
    for (int mt = 0; mt < 4; ++mt)
#pragma unroll
        for (int nt = 0; nt < 2; ++nt) acc[mt][nt] = (v8f){};

    // prologue: first k-tile into registers
    fv4 ra[4], rw[4];
    {
        const fv4* p = reinterpret_cast<const fv4*>(gA);
        ra[0] = p[0]; ra[1] = p[1]; ra[2] = p[2]; ra[3] = p[3];
        const fv4* q = reinterpret_cast<const fv4*>(gW);
        rw[0] = q[0]; rw[1] = q[1]; rw[2] = q[2]; rw[3] = q[3];
    }

    for (int k0 = 0; k0 < K; k0 += 32) {
        __syncthreads();                       // LDS free (prev reads done)
        cvt_store16(&As[srow * 40 + scol], ra);
        cvt_store16(&Ws[srow * 40 + scol], rw);
        __syncthreads();                       // LDS tile ready

        if (k0 + 32 < K) {                     // prefetch next k-tile into regs
            const fv4* p = reinterpret_cast<const fv4*>(gA + k0 + 32);
            ra[0] = p[0]; ra[1] = p[1]; ra[2] = p[2]; ra[3] = p[3];
            const fv4* q = reinterpret_cast<const fv4*>(gW + k0 + 32);
            rw[0] = q[0]; rw[1] = q[1]; rw[2] = q[2]; rw[3] = q[3];
        }

        // A fragments: lane ln -> row, contiguous K {half*16..+15}
        v16bf afr[4];
#pragma unroll
        for (int mt = 0; mt < 4; ++mt) {
            const __bf16* r = &As[(wm * 64 + mt * 16 + ln) * 40];
            afr[mt] = ld_frag2(r + half * 16, r + half * 16 + 8);
        }
        // B fragments: lane ln -> output col (row of W), contiguous K strip
        v16bf bfr[2];
#pragma unroll
        for (int nt = 0; nt < 2; ++nt) {
            const __bf16* r = &Ws[(wn * 32 + nt * 16 + ln) * 40 + half * 16];
            bfr[nt] = ld_frag32(r);
        }
#pragma unroll
        for (int mt = 0; mt < 4; ++mt)
#pragma unroll
            for (int nt = 0; nt < 2; ++nt)
                acc[mt][nt] = wmma_bf16(afr[mt], bfr[nt], acc[mt][nt]);
    }

    // ---- epilogue: C layout row = v + 8*half, col = ln ----
#pragma unroll
    for (int mt = 0; mt < 4; ++mt) {
#pragma unroll
        for (int nt = 0; nt < 2; ++nt) {
            const int col = bn + wn * 32 + nt * 16 + ln;
            const float bz = bias[col];
#pragma unroll
            for (int v = 0; v < 8; ++v) {
                const int row = bm + wm * 64 + mt * 16 + v + 8 * half;
                const float val = acc[mt][nt][v] + bz;
                if (OUT_MODE == 0) {
                    ((__bf16*)out)[(size_t)row * N + col] = (__bf16)val;
                } else if (OUT_MODE == 1) {
                    const int b = row >> 11, s = row & 2047;
                    const int h = col >> 7,  d = col & 127;
                    ((__bf16*)out)[((size_t)((b * NH + h) * DKH + d)) * SEQ + s] =
                        (__bf16)val;
                } else {
                    ((float*)out)[(size_t)row * N + col] = val;
                }
            }
        }
    }
}

// ============================================================================
// Flash attention: one block per (q-tile of 128, head, batch); 8 waves, each
// wave owns 16 q rows. K/V tiles (32 keys) are staged to LDS once per block
// with double-buffered async global->LDS copies, so WMMA fragments never wait
// on global memory. Online softmax with shfl_xor row reductions.
// Qb/Kb bf16 [tok, h*128+d]; Vt bf16 [((b*NH+h)*128+d)*SEQ + s]; O fp32.
// ============================================================================
__global__ __launch_bounds__(256) void frac_attn_kernel(
    const __bf16* __restrict__ Qb, const __bf16* __restrict__ Kb,
    const __bf16* __restrict__ Vt,
    const float* __restrict__ alpha, const float* __restrict__ beta,
    float* __restrict__ O)
{
    __shared__ __bf16 Ks[2][32 * 136];    // [key][d], stride 136 (272B)
    __shared__ __bf16 Vs[2][128 * 40];    // [d][key], stride 40 (80B)
    __shared__ __bf16 pbuf[8 * 16 * 40];  // per-wave 16x32 P tile

    const int tid  = threadIdx.x;
    const int lane = tid & 31;
    const int wave = tid >> 5;
    const int ln   = lane & 15;
    const int half = lane >> 4;

    const int qt = blockIdx.x;   // 0..15
    const int h  = blockIdx.y;   // 0..15
    const int b  = blockIdx.z;   // 0..1

    const int q0 = qt * 128 + wave * 16;
    const float scale = alpha[h] / fmaxf(beta[h], 1.0f);

    const __bf16* Kg = Kb + ((size_t)(b * SEQ)) * DM + h * DKH;         // +key*DM
    const __bf16* Vg = Vt + ((size_t)((b * NH + h) * DKH)) * SEQ;       // +d*SEQ+key

    // ---- resident Q A-fragments (16 rows x 128 d = 4 chunks of K=32) ----
    v16bf aQ[4];
    {
        const __bf16* qrow = Qb + ((size_t)(b * SEQ + q0 + ln)) * DM + h * DKH;
#pragma unroll
        for (int dc = 0; dc < 4; ++dc)
            aQ[dc] = ld_frag2(qrow + dc * 32 + half * 16,
                              qrow + dc * 32 + half * 16 + 8);
    }

    v8f o[8];
#pragma unroll
    for (int dt = 0; dt < 8; ++dt) o[dt] = (v8f){};
    float mrow[8], lrow[8];
#pragma unroll
    for (int v = 0; v < 8; ++v) { mrow[v] = -3.0e38f; lrow[v] = 0.0f; }

    __bf16* pw = &pbuf[wave * 16 * 40];

    // ---- cooperative K/V tile staging: 512+512 16B chunks, 2+2 per thread --
    auto stage_kv = [&](int kt, int buf) {
        const __bf16* kg = Kg + (size_t)kt * DM;
        const __bf16* vg = Vg + kt;
#pragma unroll
        for (int i = 0; i < 2; ++i) {                 // K tile 32x128
            const int c  = tid * 2 + i;
            const int r  = c >> 4, ch = c & 15;       // row, 8-bf16 chunk
            ASYNC_CP16(kg + (size_t)r * DM + ch * 8, &Ks[buf][r * 136 + ch * 8]);
        }
#pragma unroll
        for (int i = 0; i < 2; ++i) {                 // V tile 128x32
            const int c  = tid * 2 + i;
            const int r  = c >> 2, ch = c & 3;
            ASYNC_CP16(vg + (size_t)r * SEQ + ch * 8, &Vs[buf][r * 40 + ch * 8]);
        }
    };

    stage_kv(0, 0);
    int cur = 0;

    for (int kt = 0; kt < SEQ; kt += 32) {
        ASYNC_WAIT();          // this wave's async copies into Ks/Vs[cur] done
        __syncthreads();       // whole block: tile ready, prev compute finished

        if (kt + 32 < SEQ) stage_kv(kt + 32, cur ^ 1);   // overlap next tile

        const __bf16* kst = Ks[cur];
        const __bf16* vst = Vs[cur];

        // ---- scores S = Q K^T for 32 keys (two 16-key N tiles) ----
        v8f c0 = (v8f){}, c1 = (v8f){};
#pragma unroll
        for (int dc = 0; dc < 4; ++dc) {
            v16bf b0 = ld_frag32(&kst[(0 * 16 + ln) * 136 + dc * 32 + half * 16]);
            c0 = wmma_bf16(aQ[dc], b0, c0);
            v16bf b1 = ld_frag32(&kst[(1 * 16 + ln) * 136 + dc * 32 + half * 16]);
            c1 = wmma_bf16(aQ[dc], b1, c1);
        }

        // ---- online softmax (row = v + 8*half across its 16-lane group) ----
        float corr[8];
#pragma unroll
        for (int v = 0; v < 8; ++v) {
            const float s0 = c0[v] * scale;
            const float s1 = c1[v] * scale;
            float x = fmaxf(s0, s1);
            x = fmaxf(x, __shfl_xor(x, 1, 16));
            x = fmaxf(x, __shfl_xor(x, 2, 16));
            x = fmaxf(x, __shfl_xor(x, 4, 16));
            x = fmaxf(x, __shfl_xor(x, 8, 16));
            const float mn = fmaxf(mrow[v], x);
            corr[v] = __expf(mrow[v] - mn);
            const float p0 = __expf(s0 - mn);
            const float p1 = __expf(s1 - mn);
            float rs = p0 + p1;
            rs += __shfl_xor(rs, 1, 16);
            rs += __shfl_xor(rs, 2, 16);
            rs += __shfl_xor(rs, 4, 16);
            rs += __shfl_xor(rs, 8, 16);
            lrow[v] = lrow[v] * corr[v] + rs;
            mrow[v] = mn;
            // C layout -> LDS tile [16 rows][32 cols]
            pw[(v + 8 * half) * 40 + ln]      = (__bf16)p0;
            pw[(v + 8 * half) * 40 + 16 + ln] = (__bf16)p1;
        }
        // rescale running O
#pragma unroll
        for (int dt = 0; dt < 8; ++dt)
#pragma unroll
            for (int v = 0; v < 8; ++v) o[dt][v] *= corr[v];

        asm volatile("s_wait_dscnt 0x0" ::: "memory");
        __builtin_amdgcn_wave_barrier();

        // ---- reload P as A fragment (16x32, K = key) ----
        const __bf16* pr = &pw[ln * 40];
        v16bf aP = ld_frag2(pr + half * 16, pr + half * 16 + 8);
        __builtin_amdgcn_wave_barrier();

        // ---- O += P @ V ; B-frag column d = contiguous keys in Vs ----
#pragma unroll
        for (int dt = 0; dt < 8; ++dt) {
            v16bf bv = ld_frag32(&vst[(dt * 16 + ln) * 40 + half * 16]);
            o[dt] = wmma_bf16(aP, bv, o[dt]);
        }

        cur ^= 1;
    }

    // ---- normalize and write fp32 O [tok, h*128 + d] ----
    float inv[8];
#pragma unroll
    for (int v = 0; v < 8; ++v) inv[v] = 1.0f / lrow[v];
#pragma unroll
    for (int dt = 0; dt < 8; ++dt)
#pragma unroll
        for (int v = 0; v < 8; ++v)
            O[((size_t)(b * SEQ + q0 + v + 8 * half)) * DM + h * DKH + dt * 16 + ln] =
                o[dt][v] * inv[v];
}

// ============================================================================
extern "C" void kernel_launch(void* const* d_in, const int* in_sizes, int n_in,
                              void* d_out, int out_size, void* d_ws, size_t ws_size,
                              hipStream_t stream) {
    const float* query = (const float*)d_in[0];
    const float* key   = (const float*)d_in[1];
    const float* value = (const float*)d_in[2];
    const float* w_q   = (const float*)d_in[3];
    const float* b_q   = (const float*)d_in[4];
    const float* w_k   = (const float*)d_in[5];
    const float* b_k   = (const float*)d_in[6];
    const float* w_v   = (const float*)d_in[7];
    const float* b_v   = (const float*)d_in[8];
    const float* w_o   = (const float*)d_in[9];
    const float* b_o   = (const float*)d_in[10];
    const float* s_al  = (const float*)d_in[11];
    const float* s_be  = (const float*)d_in[12];
    float* out = (float*)d_out;

    // workspace layout
    const size_t act_bf16 = (size_t)MTOK * DM * sizeof(__bf16);   // 16.78 MB
    char* ws = (char*)d_ws;
    __bf16* Qb = (__bf16*)(ws);
    __bf16* Kb = (__bf16*)(ws + act_bf16);
    __bf16* Vt = (__bf16*)(ws + 2 * act_bf16);
    float*  Oa = (float*) (ws + 3 * act_bf16);                    // 33.55 MB fp32

    const dim3 gblk(256);
    const dim3 ggrid(DM / 128, MTOK / 128);   // (16, 32)

    gemm_xwT_kernel<0><<<ggrid, gblk, 0, stream>>>(query, w_q, b_q, Qb, MTOK, DM, DM);
    gemm_xwT_kernel<0><<<ggrid, gblk, 0, stream>>>(key,   w_k, b_k, Kb, MTOK, DM, DM);
    gemm_xwT_kernel<1><<<ggrid, gblk, 0, stream>>>(value, w_v, b_v, Vt, MTOK, DM, DM);

    const dim3 agrid(SEQ / 128, NH, BATCH);   // (16, 16, 2)
    frac_attn_kernel<<<agrid, gblk, 0, stream>>>(Qb, Kb, Vt, s_al, s_be, Oa);

    gemm_xwT_kernel<2><<<ggrid, gblk, 0, stream>>>(Oa, w_o, b_o, out, MTOK, DM, DM);
}
